// VAE_30588757082647
// MI455X (gfx1250) — compile-verified, matrix operand order checked by reading, not run on hardware
//
#include <hip/hip_runtime.h>
#include <hip/hip_bf16.h>

// ---------------------------------------------------------------------------
// Fused VAE forward ELBO for MI455X (gfx1250, wave32, WMMA).
//   elbo[b] = sum_j [ x*log(theta) + (1-x)*log1p(-theta) ]  -  KL(q||N(0,I))
// Five GEMMs fused into one persistent-tile kernel; f16 WMMA w/ f32 accum.
// Register blocking: 4M x 2N tiles per wave pass (A-fragment reuse = 2).
// ---------------------------------------------------------------------------

typedef _Float16 h4  __attribute__((ext_vector_type(4)));
typedef _Float16 h8  __attribute__((ext_vector_type(8)));
typedef _Float16 h16 __attribute__((ext_vector_type(16)));
typedef float    f8  __attribute__((ext_vector_type(8)));

#define BATCH 32768
#define DIN   1024
#define DLAT  128
#define DHID  1024
#define BM    64          // batch rows per workgroup
#define NTHREADS 256      // 8 waves (wave32)

// Build a 16-bit WMMA operand fragment from a row-major tile.
// ISA layout (16-bit A/B 16xK): lane = 16*kh + idx holds row/col `idx`,
// K-halves {kb..kb+7} and {kb+16..kb+23} where kb = k0 + kh*8.
// Caller passes p = &tile[idx * ld + k0 + kh*8]; we read two 16B chunks.
__device__ __forceinline__ h16 ldfrag(const _Float16* p) {
  h8 c0 = *(const h8*)(p);
  h8 c1 = *(const h8*)(p + 16);
  h16 r;
#pragma unroll
  for (int i = 0; i < 8; ++i) { r[i] = c0[i]; r[i + 8] = c1[i]; }
  return r;
}

__device__ __forceinline__ f8 wmma_f16(h16 a, h16 b, f8 c) {
  // (neg_a, A, neg_b, B, c_mod, C, reuse_a, reuse_b)
  return __builtin_amdgcn_wmma_f32_16x16x32_f16(false, a, false, b,
                                                (short)0, c, false, false);
}

// ---------------------------------------------------------------------------
// Weight convert + transpose:  out[n*K + k] = (f16) W[k*N + n]   (W is [K][N])
// ---------------------------------------------------------------------------
__global__ void cvt_wT_f16(const float* __restrict__ W, _Float16* __restrict__ o,
                           int K, int N) {
  long i = (long)blockIdx.x * blockDim.x + threadIdx.x;
  if (i < (long)K * N) {
    int n = (int)(i / K);
    int k = (int)(i - (long)n * K);
    o[i] = (_Float16)W[(long)k * N + n];
  }
}

// ---------------------------------------------------------------------------
// Fused VAE kernel: one workgroup = 64 batch rows through all five layers.
// Dynamic LDS (~272.5 KB -> 1 WG per WGP):
//   xa [BM][DIN] f16   (x tile, also reused in Bernoulli epilogue)
//   hb [BM][DHID] f16  (h, later overwritten with hd)
//   zt [BM][DLAT] f16  (latent z)
//   kl [BM] f32, ll [BM] f32  (per-row reductions)
// ---------------------------------------------------------------------------
__global__ void __launch_bounds__(NTHREADS, 1)
vae_fused(const float* __restrict__ x,   const float* __restrict__ eps,
          const float* __restrict__ be1, const float* __restrict__ bmu,
          const float* __restrict__ bls, const float* __restrict__ bd1,
          const float* __restrict__ bd2,
          const _Float16* __restrict__ We1T, const _Float16* __restrict__ WmuT,
          const _Float16* __restrict__ WlsT, const _Float16* __restrict__ Wd1T,
          const _Float16* __restrict__ Wd2T,
          float* __restrict__ out) {
  extern __shared__ _Float16 smem[];
  _Float16* xa = smem;                 // BM*DIN halves
  _Float16* hb = xa + BM * DIN;        // BM*DHID halves
  _Float16* zt = hb + BM * DHID;       // BM*DLAT halves
  float*    kl = (float*)(zt + BM * DLAT);
  float*    ll = kl + BM;

  const int tid  = threadIdx.x;
  const int wave = tid >> 5;
  const int lane = tid & 31;
  const int lm   = lane & 15;   // row (A) / col (B/C) index within 16
  const int lk   = lane >> 4;   // K-half selector
  const int row0 = blockIdx.x * BM;

  // zero per-row reduction buffers (kl and ll are contiguous)
  for (int i = tid; i < 2 * BM; i += NTHREADS) kl[i] = 0.0f;

  // stage x tile: global f32 -> LDS f16 (ds_store_b64 per 4 elems)
  {
    const float4* src = (const float4*)(x + (size_t)row0 * DIN);
    for (int i = tid; i < (BM * DIN) / 4; i += NTHREADS) {
      float4 v = src[i];
      h4 hv = { (_Float16)v.x, (_Float16)v.y, (_Float16)v.z, (_Float16)v.w };
      *(h4*)(xa + (size_t)i * 4) = hv;
    }
  }
  __syncthreads();

  // ------------------- GEMM1: h = relu(x @ We1 + be1) ----------------------
  // Output [BM][DHID] = 4 M-tiles x 64 N-tiles. Wave owns N-tile pairs
  // {wave+16j, wave+16j+8}, j=0..3; 4Mx2N register block per pass.
  for (int j = 0; j < 4; ++j) {
    const int n0 = (wave + 16 * j) * 16;        // second tile at n0 + 128
    f8 acc0[4] = {};
    f8 acc1[4] = {};
    const _Float16* pb0 = We1T + (size_t)(n0 + lm) * DIN + lk * 8;
    const _Float16* pb1 = pb0 + (size_t)128 * DIN;
    for (int k0 = 0; k0 < DIN; k0 += 32) {
      __builtin_prefetch(pb0 + k0 + 64, 0, 1);  // global_prefetch_b8
      __builtin_prefetch(pb1 + k0 + 64, 0, 1);
      h16 b0 = ldfrag(pb0 + k0);
      h16 b1 = ldfrag(pb1 + k0);
#pragma unroll
      for (int mt = 0; mt < 4; ++mt) {
        h16 a = ldfrag(xa + (size_t)(mt * 16 + lm) * DIN + k0 + lk * 8);
        acc0[mt] = wmma_f16(a, b0, acc0[mt]);
        acc1[mt] = wmma_f16(a, b1, acc1[mt]);
      }
    }
    const float bn0 = be1[n0 + lm];
    const float bn1 = be1[n0 + 128 + lm];
    _Float16* ph = hb + (size_t)(lk * 8) * DHID + n0 + lm;  // base for this lane
#pragma unroll
    for (int mt = 0; mt < 4; ++mt)
#pragma unroll
      for (int r = 0; r < 8; ++r) {
        ph[(size_t)(mt * 16 + r) * DHID]       = (_Float16)fmaxf(acc0[mt][r] + bn0, 0.0f);
        ph[(size_t)(mt * 16 + r) * DHID + 128] = (_Float16)fmaxf(acc1[mt][r] + bn1, 0.0f);
      }
  }
  __syncthreads();

  // ------ GEMM2/3: mu = h@Wmu+bmu, ls = h@Wls+bls; z = mu + exp(ls)*eps ----
  // Output [BM][DLAT=128] -> 8 n-tiles, one per wave; both heads per wave
  // (A-fragment reuse = 2 across the two heads).
  {
    const int n0 = wave * 16;
    f8 aMu[4] = {};
    f8 aLs[4] = {};
    const _Float16* pmu = WmuT + (size_t)(n0 + lm) * DHID + lk * 8;
    const _Float16* pls = WlsT + (size_t)(n0 + lm) * DHID + lk * 8;
    for (int k0 = 0; k0 < DHID; k0 += 32) {
      h16 bMu = ldfrag(pmu + k0);
      h16 bLs = ldfrag(pls + k0);
#pragma unroll
      for (int mt = 0; mt < 4; ++mt) {
        h16 a = ldfrag(hb + (size_t)(mt * 16 + lm) * DHID + k0 + lk * 8);
        aMu[mt] = wmma_f16(a, bMu, aMu[mt]);
        aLs[mt] = wmma_f16(a, bLs, aLs[mt]);
      }
    }
    const float bM = bmu[n0 + lm];
    const float bL = bls[n0 + lm];
#pragma unroll
    for (int mt = 0; mt < 4; ++mt)
#pragma unroll
      for (int r = 0; r < 8; ++r) {
        const int m   = mt * 16 + lk * 8 + r;   // C layout: lanes16-31 -> M+8
        const float mu  = aMu[mt][r] + bM;
        const float lsg = aLs[mt][r] + bL;
        const float ev  = eps[(size_t)(row0 + m) * DLAT + n0 + lm];
        const float zv  = mu + __expf(lsg) * ev;
        zt[(size_t)m * DLAT + n0 + lm] = (_Float16)zv;
        // KL contribution: 0.5*(exp(2ls) + mu^2 - 2ls - 1)
        const float term = 0.5f * (__expf(2.0f * lsg) + mu * mu - 2.0f * lsg - 1.0f);
        atomicAdd(&kl[m], term);                 // ds_add_f32
      }
  }
  __syncthreads();

  // ------------------- GEMM4: hd = relu(z @ Wd1 + bd1) ---------------------
  // K = 128 (4 WMMA K-steps). Overwrites hb (h is dead after GEMM2/3).
  for (int j = 0; j < 4; ++j) {
    const int n0 = (wave + 16 * j) * 16;
    f8 acc0[4] = {};
    f8 acc1[4] = {};
    const _Float16* pb0 = Wd1T + (size_t)(n0 + lm) * DLAT + lk * 8;
    const _Float16* pb1 = pb0 + (size_t)128 * DLAT;
#pragma unroll
    for (int k0 = 0; k0 < DLAT; k0 += 32) {
      h16 b0 = ldfrag(pb0 + k0);
      h16 b1 = ldfrag(pb1 + k0);
#pragma unroll
      for (int mt = 0; mt < 4; ++mt) {
        h16 a = ldfrag(zt + (size_t)(mt * 16 + lm) * DLAT + k0 + lk * 8);
        acc0[mt] = wmma_f16(a, b0, acc0[mt]);
        acc1[mt] = wmma_f16(a, b1, acc1[mt]);
      }
    }
    const float bn0 = bd1[n0 + lm];
    const float bn1 = bd1[n0 + 128 + lm];
    _Float16* ph = hb + (size_t)(lk * 8) * DHID + n0 + lm;
#pragma unroll
    for (int mt = 0; mt < 4; ++mt)
#pragma unroll
      for (int r = 0; r < 8; ++r) {
        ph[(size_t)(mt * 16 + r) * DHID]       = (_Float16)fmaxf(acc0[mt][r] + bn0, 0.0f);
        ph[(size_t)(mt * 16 + r) * DHID + 128] = (_Float16)fmaxf(acc1[mt][r] + bn1, 0.0f);
      }
  }
  __syncthreads();

  // --------- GEMM5 + Bernoulli log-lik: t = hd@Wd2 + bd2 -------------------
  // log sigmoid(t) = -log1p(exp(-t)); log(1-sigmoid(t)) = -log1p(exp(t)).
  {
    float lacc[4][8];
#pragma unroll
    for (int mt = 0; mt < 4; ++mt)
#pragma unroll
      for (int r = 0; r < 8; ++r) lacc[mt][r] = 0.0f;

    for (int j = 0; j < 4; ++j) {
      const int n0 = (wave + 16 * j) * 16;
      f8 acc0[4] = {};
      f8 acc1[4] = {};
      const _Float16* pb0 = Wd2T + (size_t)(n0 + lm) * DHID + lk * 8;
      const _Float16* pb1 = pb0 + (size_t)128 * DHID;
      for (int k0 = 0; k0 < DHID; k0 += 32) {
        __builtin_prefetch(pb0 + k0 + 64, 0, 1);
        __builtin_prefetch(pb1 + k0 + 64, 0, 1);
        h16 b0 = ldfrag(pb0 + k0);
        h16 b1 = ldfrag(pb1 + k0);
#pragma unroll
        for (int mt = 0; mt < 4; ++mt) {
          h16 a = ldfrag(hb + (size_t)(mt * 16 + lm) * DHID + k0 + lk * 8);
          acc0[mt] = wmma_f16(a, b0, acc0[mt]);
          acc1[mt] = wmma_f16(a, b1, acc1[mt]);
        }
      }
      const float bn0 = bd2[n0 + lm];
      const float bn1 = bd2[n0 + 128 + lm];
      const _Float16* px = xa + (size_t)(lk * 8) * DIN + n0 + lm;
#pragma unroll
      for (int mt = 0; mt < 4; ++mt)
#pragma unroll
        for (int r = 0; r < 8; ++r) {
          const float t0 = acc0[mt][r] + bn0;
          const float t1 = acc1[mt][r] + bn1;
          const float x0 = (float)px[(size_t)(mt * 16 + r) * DIN];        // exact 0/1
          const float x1 = (float)px[(size_t)(mt * 16 + r) * DIN + 128];
          lacc[mt][r] += x0 * (-log1pf(__expf(-t0))) +
                         (1.0f - x0) * (-log1pf(__expf(t0)));
          lacc[mt][r] += x1 * (-log1pf(__expf(-t1))) +
                         (1.0f - x1) * (-log1pf(__expf(t1)));
        }
    }
#pragma unroll
    for (int mt = 0; mt < 4; ++mt)
#pragma unroll
      for (int r = 0; r < 8; ++r) {
        const int m = mt * 16 + lk * 8 + r;
        atomicAdd(&ll[m], lacc[mt][r]);          // ds_add_f32
      }
  }
  __syncthreads();

  // elbo = log_px - kl
  for (int i = tid; i < BM; i += NTHREADS) out[row0 + i] = ll[i] - kl[i];
}

// ---------------------------------------------------------------------------
extern "C" void kernel_launch(void* const* d_in, const int* in_sizes, int n_in,
                              void* d_out, int out_size, void* d_ws, size_t ws_size,
                              hipStream_t stream) {
  const float* x   = (const float*)d_in[0];
  const float* eps = (const float*)d_in[1];
  const float* We1 = (const float*)d_in[2];
  const float* be1 = (const float*)d_in[3];
  const float* Wmu = (const float*)d_in[4];
  const float* bmu = (const float*)d_in[5];
  const float* Wls = (const float*)d_in[6];
  const float* bls = (const float*)d_in[7];
  const float* Wd1 = (const float*)d_in[8];
  const float* bd1 = (const float*)d_in[9];
  const float* Wd2 = (const float*)d_in[10];
  const float* bd2 = (const float*)d_in[11];
  float* out = (float*)d_out;
  (void)in_sizes; (void)n_in; (void)out_size; (void)ws_size;

  // f16 transposed weights in workspace: W^T is [N][K], row stride = K.
  _Float16* We1T = (_Float16*)d_ws;                       // [DHID][DIN]
  _Float16* WmuT = We1T + (size_t)DHID * DIN;             // [DLAT][DHID]
  _Float16* WlsT = WmuT + (size_t)DLAT * DHID;            // [DLAT][DHID]
  _Float16* Wd1T = WlsT + (size_t)DLAT * DHID;            // [DHID][DLAT]
  _Float16* Wd2T = Wd1T + (size_t)DHID * DLAT;            // [DIN][DHID]

  cvt_wT_f16<<<(DHID * DIN + 255) / 256, 256, 0, stream>>>(We1, We1T, DIN, DHID);
  cvt_wT_f16<<<(DLAT * DHID + 255) / 256, 256, 0, stream>>>(Wmu, WmuT, DHID, DLAT);
  cvt_wT_f16<<<(DLAT * DHID + 255) / 256, 256, 0, stream>>>(Wls, WlsT, DHID, DLAT);
  cvt_wT_f16<<<(DHID * DLAT + 255) / 256, 256, 0, stream>>>(Wd1, Wd1T, DLAT, DHID);
  cvt_wT_f16<<<(DIN * DHID + 255) / 256, 256, 0, stream>>>(Wd2, Wd2T, DHID, DIN);

  const size_t smem_bytes =
      (size_t)(BM * DIN + BM * DHID + BM * DLAT) * sizeof(_Float16) +
      (size_t)(2 * BM) * sizeof(float);   // ~272.5 KB: one WG per WGP (320 KB)

  hipFuncSetAttribute((const void*)vae_fused,
                      hipFuncAttributeMaxDynamicSharedMemorySize,
                      (int)smem_bytes);

  vae_fused<<<BATCH / BM, NTHREADS, smem_bytes, stream>>>(
      x, eps, be1, bmu, bls, bd1, bd2,
      We1T, WmuT, WlsT, Wd1T, Wd2T, out);
}
